// BiologicalGPT_52536039964871
// MI455X (gfx1250) — matrix operand order repaired and checked by reading, not compiled, not run on hardware
//
#include <hip/hip_runtime.h>
#include <hip/hip_fp16.h>

typedef __attribute__((ext_vector_type(16))) _Float16 v16h;
typedef __attribute__((ext_vector_type(8)))  float    v8f;

#define DIMC  768
#define NHEAD 12
#define HD    64
#define TT    1024
#define BBATCH 2
#define MTOK  (BBATCH*TT)       // 2048
#define VOCABN 32000
#define FFD   3072
#define QKVN  (3*DIMC)          // 2304
#define BHN   (BBATCH*NHEAD)    // 24
#define QKVSEG 1572864L         // BHN*TT*HD elements per q/k/v segment

// ---- GEMM tiling ----
#define BM 128
#define BN 128
#define BK 32
#define LSTR 40                 // LDS row stride in halfs (80B rows: 16B aligned, staggers banks)

union V8F  { v8f  v; float f[8]; };
union V16H { uint4 u[2]; v16h v; };

// ---- CDNA5 async global->LDS 16B copy (GLOBAL_LOAD_ASYNC_TO_LDS_B128, ASYNCcnt) ----
// Inline asm: VDST = 32-bit VGPR with LDS byte address (low 32 bits of flat shared ptr),
// VADDR = 64-bit global address. Avoids the builtin's address-space-typed pointer params.
__device__ __forceinline__ void async_copy16(__half* lds, const __half* g) {
  const unsigned l = (unsigned)(unsigned long long)lds;
  asm volatile("global_load_async_to_lds_b128 %0, %1, off"
               :: "v"(l), "v"(g) : "memory");
}

__device__ __forceinline__ void wait_async0() {
#if __has_builtin(__builtin_amdgcn_s_wait_asynccnt)
  __builtin_amdgcn_s_wait_asynccnt(0);
#else
  asm volatile("s_wait_asynccnt 0" ::: "memory");
#endif
}

struct GemmP {
  const __half* A;      // [M][K] row-major f16 (+ z*aZ)
  const __half* B;      // [N][K] row-major f16, i.e. B pre-transposed (+ z*bZ)
  int M, N, K;
  long aZ, bZ;          // per-blockIdx.z strides (elements)
  const float* bias;    // [N]
  const float* res;     // [M][N] f32 residual (MODE 3)
  const float* dna;     // [TT][TT] f32 attn bias (MODE 1, may be null)
  float*  outF;
  __half* outH;
  float scale;
};

// MODE: 0=QKV split  1=scores(scale+dna+causal)  2=PV->attn  3=bias+res f32  4=bias+GELU f16  5=bias f32
template <int MODE>
__global__ __launch_bounds__(256) void gemm_wmma(GemmP p) {
  __shared__ __align__(16) __half As[2][BM * LSTR];
  __shared__ __align__(16) __half Bs[2][BN * LSTR];
  const int tid  = threadIdx.x;
  const int lane = tid & 31;
  const int wave = tid >> 5;
  const int wm   = wave >> 2;          // 0..1  -> 64 rows each
  const int wn   = wave & 3;           // 0..3  -> 32 cols each
  const int m0   = blockIdx.y * BM;
  const int n0   = blockIdx.x * BN;
  const int z    = blockIdx.z;
  const __half* Ag = p.A + (long)z * p.aZ;
  const __half* Bg = p.B + (long)z * p.bZ;

  // Hoisted per-thread staging coordinates: 512 x 16B vectors per matrix, 2 per thread.
  long aOff[2], bOff[2];
  int  lofs[2];
#pragma unroll
  for (int it = 0; it < 2; ++it) {
    const int i   = tid + it * 256;
    const int row = i >> 2;
    const int ch  = i & 3;
    int gm = m0 + row; gm = gm < p.M ? gm : p.M - 1;   // clamp (OOB rows recomputed, discarded on store)
    int gn = n0 + row; gn = gn < p.N ? gn : p.N - 1;
    aOff[it] = (long)gm * p.K + ch * 8;
    bOff[it] = (long)gn * p.K + ch * 8;
    lofs[it] = row * LSTR + ch * 8;
  }

  auto stage = [&](int k0, int buf) {
#pragma unroll
    for (int it = 0; it < 2; ++it) {
      async_copy16(&As[buf][lofs[it]], Ag + aOff[it] + k0);
      async_copy16(&Bs[buf][lofs[it]], Bg + bOff[it] + k0);
    }
  };

  const v8f zacc = {0.f,0.f,0.f,0.f,0.f,0.f,0.f,0.f};
  v8f acc[4][2];
#pragma unroll
  for (int i = 0; i < 4; ++i)
#pragma unroll
    for (int j = 0; j < 2; ++j) acc[i][j] = zacc;

  const int hi = (lane >= 16) ? 1 : 0;
  const int lr = lane & 15;

  stage(0, 0);                         // prologue DMA into buffer 0
  int buf = 0;
  for (int k0 = 0; k0 < p.K; k0 += BK) {
    wait_async0();                     // my tile DMAs for `buf` have landed
    __syncthreads();                   // everyone's have landed; prev reads of buf^1 are done
    if (k0 + BK < p.K) stage(k0 + BK, buf ^ 1);   // overlap next DMA with this step's WMMAs

    v16h af[4], bf[2];
    // A fragment (16x32 f16): lanes 0-15 row M=lr hold K 0-7 & 16-23; lanes 16-31 hold K 8-15 & 24-31
#pragma unroll
    for (int i = 0; i < 4; ++i) {
      const int row = wm * 64 + i * 16 + lr;
      V16H t;
      t.u[0] = *(const uint4*)(&As[buf][row * LSTR + hi * 8]);
      t.u[1] = *(const uint4*)(&As[buf][row * LSTR + 16 + hi * 8]);
      af[i] = t.v;
    }
    // B fragment (32x16 f16): lane holds column N=lr; lanes 0-15 hold K 0-7 & 16-23; lanes 16-31 hold K 8-15 & 24-31
#pragma unroll
    for (int j = 0; j < 2; ++j) {
      const int row = wn * 32 + j * 16 + lr;
      V16H t;
      t.u[0] = *(const uint4*)(&Bs[buf][row * LSTR + hi * 8]);
      t.u[1] = *(const uint4*)(&Bs[buf][row * LSTR + 16 + hi * 8]);
      bf[j] = t.v;
    }
#pragma unroll
    for (int i = 0; i < 4; ++i)
#pragma unroll
      for (int j = 0; j < 2; ++j)
        acc[i][j] = __builtin_amdgcn_wmma_f32_16x16x32_f16(
            false, af[i], false, bf[j], (short)0, acc[i][j], false, false);
    buf ^= 1;
  }

  // Epilogue: C/D layout -> VGPR r: M = r + (lane<16?0:8), N = lane%16
#pragma unroll
  for (int i = 0; i < 4; ++i) {
#pragma unroll
    for (int j = 0; j < 2; ++j) {
      V8F t; t.v = acc[i][j];
#pragma unroll
      for (int r = 0; r < 8; ++r) {
        const int m = m0 + wm * 64 + i * 16 + hi * 8 + r;
        const int n = n0 + wn * 32 + j * 16 + lr;
        if (m >= p.M || n >= p.N) continue;
        float v = t.f[r];
        if (MODE == 0) {                       // QKV: bias, split into Q,K (row-major) / V (transposed)
          v += p.bias[n];
          const int sec = n / DIMC;
          const int rem = n - sec * DIMC;
          const int h = rem >> 6, d = rem & 63;
          const int b = m >> 10, tpos = m & (TT - 1);
          const __half hv = __float2half(v);
          if (sec == 0)
            p.outH[((long)((b * NHEAD + h) * TT + tpos)) * HD + d] = hv;
          else if (sec == 1)
            p.outH[QKVSEG + ((long)((b * NHEAD + h) * TT + tpos)) * HD + d] = hv;
          else
            p.outH[2L * QKVSEG + ((long)((b * NHEAD + h) * HD + d)) * TT + tpos] = hv;
        } else if (MODE == 1) {                // scores: scale, +dna bias, causal mask
          v *= p.scale;
          if (p.dna) v += p.dna[(long)m * TT + n];
          if (n > m) v = -3.0e38f;
          p.outF[(long)z * TT * TT + (long)m * TT + n] = v;
        } else if (MODE == 2) {                // PV: scatter into concat-heads activation
          const int b = z / NHEAD, h = z - b * NHEAD;
          p.outH[((long)(b * TT + m)) * DIMC + h * HD + n] = __float2half(v);
        } else if (MODE == 3) {                // bias + residual -> f32 stream
          v += p.bias[n] + p.res[(long)m * p.N + n];
          p.outF[(long)m * p.N + n] = v;
        } else if (MODE == 4) {                // bias + exact GELU -> f16
          v += p.bias[n];
          v = 0.5f * v * (1.0f + erff(v * 0.70710678118f));
          p.outH[(long)m * p.N + n] = __float2half(v);
        } else {                               // bias -> f32 (LM head)
          v += p.bias[n];
          p.outF[(long)m * p.N + n] = v;
        }
      }
    }
  }
}

__global__ __launch_bounds__(256) void embed_kernel(const int* __restrict__ tok,
                                                    const float* __restrict__ te,
                                                    const float* __restrict__ pe,
                                                    const float* __restrict__ he,
                                                    float* __restrict__ x) {
  const long i = (long)blockIdx.x * 256 + threadIdx.x;
  if (i >= (long)MTOK * DIMC) return;
  const int m = (int)(i / DIMC), c = (int)(i - (long)m * DIMC);
  const int t = m & (TT - 1);
  x[i] = te[(long)tok[m] * DIMC + c] + pe[(long)t * DIMC + c] + he[(long)t * DIMC + c];
}

__global__ __launch_bounds__(256) void ln_kernel(const float* __restrict__ x,
                                                 const float* __restrict__ g,
                                                 const float* __restrict__ b,
                                                 __half* __restrict__ out) {
  const long row = blockIdx.x;
  const float* xr = x + row * DIMC;
  const int tid = threadIdx.x, lane = tid & 31, wave = tid >> 5;
  float s = 0.f, s2 = 0.f;
#pragma unroll
  for (int j = 0; j < 3; ++j) { const float v = xr[tid + j * 256]; s += v; s2 += v * v; }
  for (int o = 16; o; o >>= 1) { s += __shfl_xor(s, o, 32); s2 += __shfl_xor(s2, o, 32); }
  __shared__ float r1[8], r2[8];
  if (lane == 0) { r1[wave] = s; r2[wave] = s2; }
  __syncthreads();
  s = 0.f; s2 = 0.f;
#pragma unroll
  for (int w = 0; w < 8; ++w) { s += r1[w]; s2 += r2[w]; }
  const float mean = s * (1.0f / DIMC);
  const float var  = s2 * (1.0f / DIMC) - mean * mean;
  const float inv  = rsqrtf(var + 1e-5f);
#pragma unroll
  for (int j = 0; j < 3; ++j) {
    const int c = tid + j * 256;
    out[row * DIMC + c] = __float2half((xr[c] - mean) * inv * g[c] + b[c]);
  }
}

__global__ __launch_bounds__(256) void softmax_kernel(const float* __restrict__ scores,
                                                      __half* __restrict__ P) {
  const long row = blockIdx.x;                 // bh*TT + q
  const int qpos = (int)(row & (TT - 1));
  const float* sr = scores + row * TT;
  __half* pr = P + row * TT;
  const int tid = threadIdx.x, lane = tid & 31, wave = tid >> 5;
  __shared__ float rmax[8], rsum[8];
  float e[4];
  float mx = -3.0e38f;
#pragma unroll
  for (int j = 0; j < 4; ++j) {
    const int c = tid + j * 256;
    const float v = (c <= qpos) ? sr[c] : -3.0e38f;
    e[j] = v;
    mx = fmaxf(mx, v);
  }
  for (int o = 16; o; o >>= 1) mx = fmaxf(mx, __shfl_xor(mx, o, 32));
  if (lane == 0) rmax[wave] = mx;
  __syncthreads();
  mx = rmax[0];
#pragma unroll
  for (int w = 1; w < 8; ++w) mx = fmaxf(mx, rmax[w]);
  float sum = 0.f;
#pragma unroll
  for (int j = 0; j < 4; ++j) {
    const int c = tid + j * 256;
    const float ev = (c <= qpos) ? __expf(e[j] - mx) : 0.f;
    e[j] = ev; sum += ev;
  }
  for (int o = 16; o; o >>= 1) sum += __shfl_xor(sum, o, 32);
  if (lane == 0) rsum[wave] = sum;
  __syncthreads();
  sum = 0.f;
#pragma unroll
  for (int w = 0; w < 8; ++w) sum += rsum[w];
  const float inv = 1.0f / sum;
#pragma unroll
  for (int j = 0; j < 4; ++j) pr[tid + j * 256] = __float2half(e[j] * inv);
}

// Wt[n][k] = (f16) W[k][n]   (W is [K][N] f32)
__global__ __launch_bounds__(256) void transpose_cvt(const float* __restrict__ W,
                                                     __half* __restrict__ Wt, int K, int N) {
  const long i = (long)blockIdx.x * 256 + threadIdx.x;
  if (i >= (long)K * N) return;
  const long n = i / K, k = i - n * K;
  Wt[i] = __float2half(W[k * N + n]);
}

extern "C" void kernel_launch(void* const* d_in, const int* in_sizes, int n_in,
                              void* d_out, int out_size, void* d_ws, size_t ws_size,
                              hipStream_t stream) {
  (void)in_sizes; (void)n_in; (void)out_size; (void)ws_size;
  const int*   tokens    = (const int*)d_in[0];
  const float* token_emb = (const float*)d_in[1];
  const float* pos_emb   = (const float*)d_in[2];
  const float* helical   = (const float*)d_in[3];
  const float* ln_f_g    = (const float*)d_in[4];
  const float* ln_f_b    = (const float*)d_in[5];
  const float* lm_w      = (const float*)d_in[6];
  const float* lm_b      = (const float*)d_in[7];

  struct BlkPtrs {
    const float *ln1_g, *ln1_b, *ln2_g, *ln2_b, *qkv_w, *qkv_b, *out_w, *out_b,
                *ff1_w, *ff1_b, *ff2_w, *ff2_b, *dna;
  } blk[6];
  int idx = 8;
  for (int i = 0; i < 6; ++i) {
    blk[i].ln1_g = (const float*)d_in[idx++];
    blk[i].ln1_b = (const float*)d_in[idx++];
    blk[i].ln2_g = (const float*)d_in[idx++];
    blk[i].ln2_b = (const float*)d_in[idx++];
    blk[i].qkv_w = (const float*)d_in[idx++];
    blk[i].qkv_b = (const float*)d_in[idx++];
    blk[i].out_w = (const float*)d_in[idx++];
    blk[i].out_b = (const float*)d_in[idx++];
    blk[i].ff1_w = (const float*)d_in[idx++];
    blk[i].ff1_b = (const float*)d_in[idx++];
    blk[i].ff2_w = (const float*)d_in[idx++];
    blk[i].ff2_b = (const float*)d_in[idx++];
    if ((i & 1) == 0) { blk[i].dna = (const float*)d_in[idx++]; }
    else { blk[i].dna = nullptr; idx += 2; }  // motif bias is per-query => softmax-invariant, skip
  }

  char* ws = (char*)d_ws;
  float*  x      = (float*) (ws + 0);          //  6,291,456 B  residual stream f32
  __half* xn     = (__half*)(ws + 6291456);    //  3,145,728 B  layernormed f16
  __half* q      = (__half*)(ws + 9437184);    //  3*QKVSEG halfs: q, k, vT (9,437,184 B)
  __half* attn   = (__half*)(ws + 18874368);   //  3,145,728 B
  __half* hbuf   = (__half*)(ws + 22020096);   // 12,582,912 B
  __half* wbuf   = (__half*)(ws + 34603008);   //  4,718,592 B transposed-weight scratch
  float*  scores = (float*) (ws + 39321600);   // 100,663,296 B
  __half* P      = (__half*)(ws + 139984896);  // 50,331,648 B
  __half* lmT    = (__half*)(ws + 39321600);   // 49,152,000 B, overlaps dead scores region

  const dim3 b256(256);
  { const long tot = (long)MTOK * DIMC;
    embed_kernel<<<dim3((unsigned)((tot + 255) / 256)), b256, 0, stream>>>(tokens, token_emb, pos_emb, helical, x); }

  for (int i = 0; i < 6; ++i) {
    ln_kernel<<<dim3(MTOK), b256, 0, stream>>>(x, blk[i].ln1_g, blk[i].ln1_b, xn);

    { const long tot = (long)DIMC * QKVN;
      transpose_cvt<<<dim3((unsigned)((tot + 255) / 256)), b256, 0, stream>>>(blk[i].qkv_w, wbuf, DIMC, QKVN); }
    GemmP pq{}; pq.A = xn; pq.B = wbuf; pq.M = MTOK; pq.N = QKVN; pq.K = DIMC;
    pq.bias = blk[i].qkv_b; pq.outH = q;
    gemm_wmma<0><<<dim3(QKVN / BN, MTOK / BM, 1), b256, 0, stream>>>(pq);

    GemmP ps{}; ps.A = q; ps.B = q + QKVSEG; ps.M = TT; ps.N = TT; ps.K = HD;
    ps.aZ = (long)TT * HD; ps.bZ = (long)TT * HD; ps.dna = blk[i].dna;
    ps.outF = scores; ps.scale = 0.125f;   // 1/sqrt(64)
    gemm_wmma<1><<<dim3(TT / BN, TT / BM, BHN), b256, 0, stream>>>(ps);

    softmax_kernel<<<dim3(BHN * TT), b256, 0, stream>>>(scores, P);

    GemmP pv{}; pv.A = P; pv.B = q + 2L * QKVSEG; pv.M = TT; pv.N = HD; pv.K = TT;
    pv.aZ = (long)TT * TT; pv.bZ = (long)HD * TT; pv.outH = attn;
    gemm_wmma<2><<<dim3(1, TT / BM, BHN), b256, 0, stream>>>(pv);

    { const long tot = (long)DIMC * DIMC;
      transpose_cvt<<<dim3((unsigned)((tot + 255) / 256)), b256, 0, stream>>>(blk[i].out_w, wbuf, DIMC, DIMC); }
    GemmP po{}; po.A = attn; po.B = wbuf; po.M = MTOK; po.N = DIMC; po.K = DIMC;
    po.bias = blk[i].out_b; po.res = x; po.outF = x;
    gemm_wmma<3><<<dim3(DIMC / BN, MTOK / BM, 1), b256, 0, stream>>>(po);

    ln_kernel<<<dim3(MTOK), b256, 0, stream>>>(x, blk[i].ln2_g, blk[i].ln2_b, xn);

    { const long tot = (long)DIMC * FFD;
      transpose_cvt<<<dim3((unsigned)((tot + 255) / 256)), b256, 0, stream>>>(blk[i].ff1_w, wbuf, DIMC, FFD); }
    GemmP f1{}; f1.A = xn; f1.B = wbuf; f1.M = MTOK; f1.N = FFD; f1.K = DIMC;
    f1.bias = blk[i].ff1_b; f1.outH = hbuf;
    gemm_wmma<4><<<dim3(FFD / BN, MTOK / BM, 1), b256, 0, stream>>>(f1);

    { const long tot = (long)FFD * DIMC;
      transpose_cvt<<<dim3((unsigned)((tot + 255) / 256)), b256, 0, stream>>>(blk[i].ff2_w, wbuf, FFD, DIMC); }
    GemmP f2{}; f2.A = hbuf; f2.B = wbuf; f2.M = MTOK; f2.N = DIMC; f2.K = FFD;
    f2.bias = blk[i].ff2_b; f2.res = x; f2.outF = x;
    gemm_wmma<3><<<dim3(DIMC / BN, MTOK / BM, 1), b256, 0, stream>>>(f2);
  }

  ln_kernel<<<dim3(MTOK), b256, 0, stream>>>(x, ln_f_g, ln_f_b, xn);
  { const long tot = (long)DIMC * VOCABN;
    transpose_cvt<<<dim3((unsigned)((tot + 255) / 256)), b256, 0, stream>>>(lm_w, lmT, DIMC, VOCABN); }
  GemmP lm{}; lm.A = xn; lm.B = lmT; lm.M = MTOK; lm.N = VOCABN; lm.K = DIMC;
  lm.bias = lm_b; lm.outF = (float*)d_out;
  gemm_wmma<5><<<dim3(VOCABN / BN, MTOK / BM, 1), b256, 0, stream>>>(lm);
}
